// VectorQuantizer_29300266893695
// MI455X (gfx1250) — matrix operand order repaired and checked by reading, not compiled, not run on hardware
//
#include <hip/hip_runtime.h>
#include <hip/hip_bf16.h>

// ---------------------------------------------------------------------------
// VQ-VAE vector quantizer for MI455X (gfx1250, wave32, WMMA).
// distances = ||x||^2 - 2 x.c^T + ||c||^2 ; argmin over V only needs
// (||c||^2 - 2 x.c^T). GEMM done with v_wmma_f32_16x16x32_bf16.
// Argmin bookkeeping packs the 10-bit code id into the score's low mantissa
// bits so the running reduction is a single v_and_or_b32 + v_min_num_f32.
// loss = (1 + 0.25) * mean((q - x)^2)   (both sub-losses are numerically equal)
// ---------------------------------------------------------------------------

#define VQ_N 65536   // B*M
#define VQ_H 64
#define VQ_V 1024

typedef __attribute__((ext_vector_type(16))) __bf16        v16bf;
typedef __attribute__((ext_vector_type(8)))  float         v8f;
typedef __attribute__((ext_vector_type(4)))  unsigned int  u32x4;
typedef __attribute__((ext_vector_type(4)))  float         f32x4;
typedef __attribute__((ext_vector_type(4)))  __bf16        bf16x4;

union FragBF {
    v16bf v;
    u32x4 u[2];
};

// ---- pass 1a: x (f32) -> bf16 workspace, 4 elements / thread ---------------
__global__ __launch_bounds__(256) void k_cvt_x(const float* __restrict__ x,
                                               __bf16* __restrict__ xb) {
    int i = (blockIdx.x * 256 + threadIdx.x) * 4;     // NH divisible by 1024
    f32x4 v = *(const f32x4*)(x + i);
    bf16x4 o;
    o[0] = (__bf16)v[0];
    o[1] = (__bf16)v[1];
    o[2] = (__bf16)v[2];
    o[3] = (__bf16)v[3];
    *(bf16x4*)(xb + i) = o;
}

// ---- pass 1b: codebook -> bf16 + per-code squared norms; zero loss slot ----
__global__ __launch_bounds__(256) void k_prep_cb(const float* __restrict__ cb,
                                                 __bf16* __restrict__ cbh,
                                                 float* __restrict__ cnorm,
                                                 float* __restrict__ loss) {
    int row = blockIdx.x * 256 + threadIdx.x;
    if (row == 0) *loss = 0.0f;
    if (row < VQ_V) {
        float s = 0.0f;
        #pragma unroll 8
        for (int d = 0; d < VQ_H; ++d) {
            float v = cb[row * VQ_H + d];
            s += v * v;
            cbh[row * VQ_H + d] = (__bf16)v;
        }
        cnorm[row] = s;
    }
}

// ---- pass 2: WMMA distance GEMM + running argmin ---------------------------
// One wave handles 16 rows of x. A (16x64 bf16) resident in VGPRs.
// Loop 64 tiles of 16 codes: B frag loads + 2x wmma (K=64=2x32).
//
// A frag (16x32, v16bf): lane L -> row r=L&15, kh=L>>4.
//   halfs 0-7  = K[kh*8   .. kh*8+7]   halfs 8-15 = K[16+kh*8 .. 16+kh*8+7]
// B frag (32x16, v16bf): lane L -> col n=L&15, kh=L>>4.
//   halfs 0-7  = K[kh*8   .. kh*8+7]   halfs 8-15 = K[16+kh*8 .. 16+kh*8+7]
// C/D (v8f): lane L -> col n=L&15 ; VGPR j -> row m = j + 8*(L>>4).
__global__ __launch_bounds__(256) void k_vq_argmin(
    const __bf16* __restrict__ xb,     // [N,64] bf16
    const __bf16* __restrict__ cbh,    // [V,64] bf16
    const float*  __restrict__ cnorm,  // [V]
    int*          __restrict__ out_idx)
{
    const int lane    = threadIdx.x & 31;
    const int wave    = threadIdx.x >> 5;
    const int rowBase = (blockIdx.x * 8 + wave) * 16;
    const int r       = lane & 15;
    const int kh      = lane >> 4;

    // Resident A fragments for K-chunks 0 (dims 0-31) and 1 (dims 32-63).
    const __bf16* xrow = xb + (size_t)(rowBase + r) * VQ_H;
    FragBF a0, a1;
    a0.u[0] = *(const u32x4*)(xrow +  0 + kh * 8);
    a0.u[1] = *(const u32x4*)(xrow + 16 + kh * 8);
    a1.u[0] = *(const u32x4*)(xrow + 32 + kh * 8);
    a1.u[1] = *(const u32x4*)(xrow + 48 + kh * 8);

    // Running min of composite scores: bits = (score & ~0x3FF) | code_id.
    float minv[8];
    #pragma unroll
    for (int j = 0; j < 8; ++j) minv[j] = 3.402823466e38f;

    #pragma unroll 2
    for (int t = 0; t < VQ_V / 16; ++t) {
        const int n = t * 16 + r;                     // code column this lane owns
        const __bf16* crow = cbh + (size_t)n * VQ_H;
        FragBF b0, b1;
        b0.u[0] = *(const u32x4*)(crow +  0 + kh * 8);
        b0.u[1] = *(const u32x4*)(crow + 16 + kh * 8);
        b1.u[0] = *(const u32x4*)(crow + 32 + kh * 8);
        b1.u[1] = *(const u32x4*)(crow + 48 + kh * 8);
        const float cn = cnorm[n];

        v8f c = {};
        c = __builtin_amdgcn_wmma_f32_16x16x32_bf16(false, a0.v, false, b0.v,
                                                    (short)0, c, false, false);
        c = __builtin_amdgcn_wmma_f32_16x16x32_bf16(false, a1.v, false, b1.v,
                                                    (short)0, c, false, false);

        #pragma unroll
        for (int j = 0; j < 8; ++j) {
            float s = fmaf(-2.0f, c[j], cn);          // ||c||^2 - 2 x.c
            // v_and_or_b32: clear 10 mantissa LSBs, insert code id n
            float sc = __uint_as_float(
                (__float_as_uint(s) & 0xFFFFFC00u) | (unsigned)n);
            minv[j] = fminf(minv[j], sc);             // v_min_num_f32
        }
    }

    // Cross-lane min within each 16-lane half (xor 1,2,4,8 stays in half).
    #pragma unroll
    for (int j = 0; j < 8; ++j) {
        float m = minv[j];
        #pragma unroll
        for (int d = 1; d <= 8; d <<= 1)
            m = fminf(m, __shfl_xor(m, d, 32));
        minv[j] = m;
    }

    if (lane == 0 || lane == 16) {        // rows m = j + 8*kh
        const int base = rowBase + kh * 8;
        #pragma unroll
        for (int j = 0; j < 8; ++j)
            out_idx[base + j] = (int)(__float_as_uint(minv[j]) & 0x3FFu);
    }
}

// ---- pass 3: gather quantized rows + loss reduction ------------------------
__global__ __launch_bounds__(256) void k_gather_loss(
    const float* __restrict__ x,       // [N,64] f32
    const float* __restrict__ cb,      // [V,64] f32
    const int*   __restrict__ idx,     // [N]
    float*       __restrict__ outq,    // [N,64]
    float*       __restrict__ loss)
{
    __shared__ float red[8];
    const int t   = blockIdx.x * 256 + threadIdx.x;   // < N*H
    const int row = t >> 6;
    const int d   = t & 63;
    const int id  = idx[row];
    const float q = cb[id * VQ_H + d];
    outq[t] = q;
    float diff = q - x[t];
    float p = diff * diff;

    #pragma unroll
    for (int m = 16; m >= 1; m >>= 1) p += __shfl_xor(p, m, 32);
    const int lane = threadIdx.x & 31, w = threadIdx.x >> 5;
    if (lane == 0) red[w] = p;
    __syncthreads();
    if (threadIdx.x == 0) {
        float s = 0.0f;
        #pragma unroll
        for (int i = 0; i < 8; ++i) s += red[i];
        // loss = 1.25 * mean((q-x)^2)
        atomicAdd(loss, s * (1.25f / (float)(VQ_N * VQ_H)));
    }
}

// ---------------------------------------------------------------------------
extern "C" void kernel_launch(void* const* d_in, const int* in_sizes, int n_in,
                              void* d_out, int out_size, void* d_ws, size_t ws_size,
                              hipStream_t stream) {
    (void)in_sizes; (void)n_in; (void)out_size; (void)ws_size;

    const float* x  = (const float*)d_in[0];   // [B,M,H] = [N,64]
    const float* cb = (const float*)d_in[1];   // [V,64]

    // d_out: quantized [N*64] f32 | loss [1] f32 | indices [N] int32-bits
    float* outq  = (float*)d_out;
    float* loss  = outq + (size_t)VQ_N * VQ_H;
    int*   oidx  = (int*)(outq + (size_t)VQ_N * VQ_H + 1);

    // workspace: x_bf16 | cb_bf16 | cnorm
    char*   w     = (char*)d_ws;
    __bf16* xb    = (__bf16*)w;                                    // 8 MB
    __bf16* cbh   = (__bf16*)(w + (size_t)VQ_N * VQ_H * 2);        // 128 KB
    float*  cnorm = (float*)(w + (size_t)VQ_N * VQ_H * 2
                               + (size_t)VQ_V * VQ_H * 2);         // 4 KB

    const int NH = VQ_N * VQ_H;
    k_cvt_x   <<<NH / 1024, 256, 0, stream>>>(x, xb);
    k_prep_cb <<<VQ_V / 256, 256, 0, stream>>>(cb, cbh, cnorm, loss);

    // 16 rows/wave, 8 waves/block -> 128 rows/block -> 512 blocks
    k_vq_argmin<<<VQ_N / 128, 256, 0, stream>>>(xb, cbh, cnorm, oidx);

    k_gather_loss<<<NH / 256, 256, 0, stream>>>(x, cb, oidx, outq, loss);
}
